// RNN_77429670412989
// MI455X (gfx1250) — compile-verified, matrix operand order checked by reading, not run on hardware
//
#include <hip/hip_runtime.h>
#include <math.h>

// Problem constants (match reference)
#define SEQ   2048
#define BATCH 128
#define IN_D  256
#define HID   512
#define OUT_D 256

// Launch geometry for the persistent recurrent kernel
#define NUM_WG       32
#define TPB          256          // 8 waves of 32 (wave32)
#define WAVES_PER_WG 8

typedef __attribute__((ext_vector_type(16))) _Float16 v16h;
typedef __attribute__((ext_vector_type(8)))  float    v8f;

// ---------------- workspace layout (bytes) ----------------
#define OFF_WHH   0u                                   // HID*HID  halfs (512 KB)
#define OFF_WIH   (OFF_WHH + HID*HID*2u)               // HID*IN_D halfs (256 KB)
#define OFF_WFC   (OFF_WIH + HID*IN_D*2u)              // OUT*HID  halfs (256 KB)
#define OFF_HA    (OFF_WFC + OUT_D*HID*2u)             // BATCH*HID halfs (128 KB)
#define OFF_HB    (OFF_HA + BATCH*HID*2u)              // BATCH*HID halfs (128 KB)
#define OFF_BIAS  (OFF_HB + BATCH*HID*2u)              // HID f32
#define OFF_CTR   (OFF_BIAS + 4096u)                   // barrier counter (u32)
#define OFF_XP    (2u << 20)                           // optional x_proj, [s][h][b] f32 (512 MB)

// K mapping of the 16-bit WMMA A/B operand layout (CDNA5 ISA §7.12.2):
//   half-slot s (0..15) of lane L -> k = (s<8 ? s : s+8) + (L<16 ? 0 : 8) within a
//   K=32 fragment; lane%16 is the M row (A) / N column (B).

__device__ __forceinline__ float fast_tanh(float v) {
#if __has_builtin(__builtin_amdgcn_tanhf)
    return __builtin_amdgcn_tanhf(v);   // single V_TANH_F32 TRANS op on gfx1250
#else
    return tanhf(v);
#endif
}

// CDNA5 async global -> LDS copy (ASYNCcnt-tracked), CDNA5 ISA.
__device__ __forceinline__ void async_copy_b128(unsigned lds_off, const void* gaddr) {
    asm volatile("global_load_async_to_lds_b128 %0, %1, off"
                 :: "v"(lds_off), "v"((unsigned long long)(uintptr_t)gaddr)
                 : "memory");
}
__device__ __forceinline__ void wait_async0() {
    asm volatile("s_wait_asynccnt 0x0" ::: "memory");
}
// Stage `halfs` f16 elements (multiple of 8*TPB) from global into LDS, whole block.
__device__ __forceinline__ void stage_to_lds(_Float16* dst_lds, const _Float16* src, int halfs) {
    for (int i = threadIdx.x * 8; i < halfs; i += TPB * 8) {
        async_copy_b128((unsigned)(uintptr_t)(dst_lds + i), src + i);
    }
    wait_async0();
    __syncthreads();
}

// Pack a [N,K] f32 row-major weight (used as B^T) into f16 fragment order:
// frag = n_tile*(K/32)+f, 512 halfs/frag, element (lane,s) =
// W[n_tile*16 + lane%16][f*32 + kmap(lane,s)]; main loop loads 32 contiguous B/lane.
__global__ void pack_b_frags(const float* __restrict__ W,
                             _Float16* __restrict__ dst, int N, int K) {
    int e = blockIdx.x * blockDim.x + threadIdx.x;
    if (e >= N * K) return;
    int frag = e >> 9;
    int r    = e & 511;
    int lane = r >> 4;
    int s    = r & 15;
    int kf   = K >> 5;
    int t    = frag / kf;
    int f    = frag - t * kf;
    int n    = t * 16 + (lane & 15);
    int k    = f * 32 + ((s < 8 ? s : s + 8) + ((lane < 16) ? 0 : 8));
    dst[e] = (_Float16)W[(size_t)n * K + k];
}

// Zero h0, fuse biases, reset the grid-barrier counter (every launch).
__global__ void init_state(unsigned* __restrict__ h0_u32,
                           float* __restrict__ bias,
                           const float* __restrict__ b_ih,
                           const float* __restrict__ b_hh,
                           unsigned* __restrict__ ctr) {
    int i = blockIdx.x * blockDim.x + threadIdx.x;
    if (i < BATCH * HID / 2) h0_u32[i] = 0u;
    if (i < HID) bias[i] = b_ih[i] + b_hh[i];
    if (i == 0) *ctr = 0u;
}

__device__ __forceinline__ void grid_barrier(unsigned* ctr, unsigned target) {
    __syncthreads();
    if (threadIdx.x == 0) {
        __hip_atomic_fetch_add(ctr, 1u, __ATOMIC_RELEASE, __HIP_MEMORY_SCOPE_AGENT);
        while (__hip_atomic_load(ctr, __ATOMIC_ACQUIRE, __HIP_MEMORY_SCOPE_AGENT) < target) {
            __builtin_amdgcn_s_sleep(1);
        }
    }
    __syncthreads();
}

__device__ __forceinline__ v16h load_frag_pair(const _Float16* base, int stride2) {
    // two 16-byte chunks: halves [0..7] at base, [8..15] at base+stride2
    v16h b;
    *(float4*)&b         = *(const float4*)(base);
    *(((float4*)&b) + 1) = *(const float4*)(base + stride2);
    return b;
}

// Parallel precompute of x_proj (bias folded), stored [s][h][b] f32.
// One wave per 16x16 tile of the [SEQ*BATCH, HID] output.
__global__ void __launch_bounds__(TPB)
xproj_gemm(const float* __restrict__ x, const _Float16* __restrict__ Wih,
           const float* __restrict__ bias, float* __restrict__ xp) {
    const int lane  = threadIdx.x & 31;
    const int wid   = threadIdx.x >> 5;
    const int gw    = blockIdx.x * WAVES_PER_WG + wid;
    const int mtile = gw >> 5;          // 0 .. SEQ*BATCH/16-1
    const int nt    = gw & 31;          // 0..31
    const int lhalf = lane & 15;
    const int lhi   = lane >> 4;
    const int s_idx = mtile >> 3;
    const int mt    = mtile & 7;
    const int ncol  = nt * 16 + lhalf;

    v8f acc;
    const float bv = bias[ncol];
    #pragma unroll
    for (int d = 0; d < 8; ++d) acc[d] = bv;

    const float* __restrict__ xrow = x + ((size_t)s_idx * BATCH + mt * 16 + lhalf) * IN_D;
    #pragma unroll 2
    for (int f = 0; f < IN_D / 32; ++f) {
        const int k0 = f * 32 + (lhi ? 8 : 0);
        float4 x0 = *(const float4*)(xrow + k0);
        float4 x1 = *(const float4*)(xrow + k0 + 4);
        float4 x2 = *(const float4*)(xrow + k0 + 16);
        float4 x3 = *(const float4*)(xrow + k0 + 20);
        v16h a;
        a[0]=(_Float16)x0.x; a[1]=(_Float16)x0.y; a[2]=(_Float16)x0.z; a[3]=(_Float16)x0.w;
        a[4]=(_Float16)x1.x; a[5]=(_Float16)x1.y; a[6]=(_Float16)x1.z; a[7]=(_Float16)x1.w;
        a[8]=(_Float16)x2.x; a[9]=(_Float16)x2.y; a[10]=(_Float16)x2.z; a[11]=(_Float16)x2.w;
        a[12]=(_Float16)x3.x; a[13]=(_Float16)x3.y; a[14]=(_Float16)x3.z; a[15]=(_Float16)x3.w;
        v16h b = load_frag_pair(Wih + ((size_t)(nt * (IN_D / 32) + f) << 9) + lane * 16, 8);
        acc = __builtin_amdgcn_wmma_f32_16x16x32_f16(false, a, false, b,
                                                     (short)0, acc, false, false);
    }
    // D layout: lane holds column ncol, rows mt*16 + 8*lhi + d (d consecutive)
    float* dst = xp + ((size_t)s_idx * HID + ncol) * BATCH + mt * 16 + 8 * lhi;
    *(v8f*)dst = acc;
}

// Persistent recurrent kernel: 256 waves, each owns one 16x16 tile of h_new.
// Each workgroup = one n_tile; its B fragments live in LDS.
template <bool USE_XP>
__global__ void __launch_bounds__(TPB)
rnn_persistent(const float* __restrict__ x,
               const float* __restrict__ xp,          // x_proj [s][h][b] (if USE_XP)
               const float* __restrict__ bias,        // b_ih+b_hh fused (f32)
               const _Float16* __restrict__ Whh,      // packed frags, K=HID
               const _Float16* __restrict__ Wih,      // packed frags, K=IN_D
               const _Float16* __restrict__ Wfc,      // packed frags, K=HID
               const float* __restrict__ bfc,
               _Float16* __restrict__ hA,
               _Float16* __restrict__ hB,
               float* __restrict__ out,
               unsigned* __restrict__ ctr) {
    __shared__ __align__(16) _Float16 sWhh[16 * 512];  // 16 KB: this WG's W_hh slice
    __shared__ __align__(16) _Float16 sWih[8 * 512];   //  8 KB: this WG's W_ih slice

    const int lane   = threadIdx.x & 31;
    const int wid    = threadIdx.x >> 5;
    const int gw     = blockIdx.x * WAVES_PER_WG + wid;
    const int m_tile = gw & 7;
    const int n_tile = gw >> 3;           // == blockIdx.x
    const int lhalf  = lane & 15;
    const int lhi    = lane >> 4;

    const int arow = m_tile * 16 + lhalf;
    const int ncol = n_tile * 16 + lhalf;
    const float bv = bias[ncol];

    // one-time async stage of loop-invariant weight slices into LDS
    stage_to_lds(sWhh, Whh + ((size_t)n_tile * 16 << 9), 16 * 512);
    if (!USE_XP) stage_to_lds(sWih, Wih + ((size_t)n_tile * 8 << 9), 8 * 512);

    _Float16* hbuf[2] = { hA, hB };

    for (int s = 0; s < SEQ; ++s) {
        const _Float16* __restrict__ hin  = hbuf[s & 1];
        _Float16* __restrict__ hout       = hbuf[(s + 1) & 1];

        v8f acc;
        if (USE_XP) {
            acc = *(const v8f*)(xp + ((size_t)s * HID + ncol) * BATCH + m_tile * 16 + 8 * lhi);
        } else {
            #pragma unroll
            for (int d = 0; d < 8; ++d) acc[d] = bv;
            const float* __restrict__ xrow = x + ((size_t)s * BATCH + arow) * IN_D;
            #pragma unroll 2
            for (int f = 0; f < IN_D / 32; ++f) {
                const int k0 = f * 32 + (lhi ? 8 : 0);
                float4 x0 = *(const float4*)(xrow + k0);
                float4 x1 = *(const float4*)(xrow + k0 + 4);
                float4 x2 = *(const float4*)(xrow + k0 + 16);
                float4 x3 = *(const float4*)(xrow + k0 + 20);
                v16h a;
                a[0]=(_Float16)x0.x; a[1]=(_Float16)x0.y; a[2]=(_Float16)x0.z; a[3]=(_Float16)x0.w;
                a[4]=(_Float16)x1.x; a[5]=(_Float16)x1.y; a[6]=(_Float16)x1.z; a[7]=(_Float16)x1.w;
                a[8]=(_Float16)x2.x; a[9]=(_Float16)x2.y; a[10]=(_Float16)x2.z; a[11]=(_Float16)x2.w;
                a[12]=(_Float16)x3.x; a[13]=(_Float16)x3.y; a[14]=(_Float16)x3.z; a[15]=(_Float16)x3.w;
                v16h b = load_frag_pair(&sWih[(f << 9) + lane * 16], 8);   // ds_load
                acc = __builtin_amdgcn_wmma_f32_16x16x32_f16(false, a, false, b,
                                                             (short)0, acc, false, false);
            }
        }

        // recurrence: h (f16, global/L2) x W_hh^T (LDS)
        const _Float16* __restrict__ hrow = hin + (size_t)arow * HID;
        #pragma unroll 4
        for (int f = 0; f < HID / 32; ++f) {
            const int k0 = f * 32 + (lhi ? 8 : 0);
            v16h a = load_frag_pair(hrow + k0, 16);                        // global
            v16h b = load_frag_pair(&sWhh[(f << 9) + lane * 16], 8);       // ds_load
            acc = __builtin_amdgcn_wmma_f32_16x16x32_f16(false, a, false, b,
                                                         (short)0, acc, false, false);
        }

        #pragma unroll
        for (int d = 0; d < 8; ++d) {
            int row = m_tile * 16 + d + 8 * lhi;
            hout[(size_t)row * HID + ncol] = (_Float16)fast_tanh(acc[d]);
        }

        if (s + 1 < SEQ) {
            if (USE_XP)
                __builtin_prefetch(xp + ((size_t)(s + 1) * HID + ncol) * BATCH, 0, 1);
            else
                __builtin_prefetch(x + ((size_t)(s + 1) * BATCH + arow) * IN_D, 0, 1);
        }

        grid_barrier(ctr, (unsigned)(NUM_WG * (s + 1)));
    }

    // ---- final FC on h_last (in hA): [128,512] @ W_fc^T. Blocks 0..15 only. ----
    if (blockIdx.x < (OUT_D / 16)) {
        const int ot = blockIdx.x;
        // re-stage this block's W_fc slice into the (now free) sWhh buffer
        stage_to_lds(sWhh, Wfc + ((size_t)ot * 16 << 9), 16 * 512);

        const int mt   = m_tile;
        const int ocol = ot * 16 + lhalf;
        v8f acc;
        const float bo = bfc[ocol];
        #pragma unroll
        for (int d = 0; d < 8; ++d) acc[d] = bo;

        const _Float16* __restrict__ hrow = hA + (size_t)(mt * 16 + lhalf) * HID;
        #pragma unroll 4
        for (int f = 0; f < HID / 32; ++f) {
            const int k0 = f * 32 + (lhi ? 8 : 0);
            v16h a = load_frag_pair(hrow + k0, 16);
            v16h b = load_frag_pair(&sWhh[(f << 9) + lane * 16], 8);
            acc = __builtin_amdgcn_wmma_f32_16x16x32_f16(false, a, false, b,
                                                         (short)0, acc, false, false);
        }
        #pragma unroll
        for (int d = 0; d < 8; ++d) {
            int row = mt * 16 + d + 8 * lhi;
            out[(size_t)row * OUT_D + ocol] = acc[d];
        }
    }
}

extern "C" void kernel_launch(void* const* d_in, const int* in_sizes, int n_in,
                              void* d_out, int out_size, void* d_ws, size_t ws_size,
                              hipStream_t stream) {
    const float* x    = (const float*)d_in[0];
    const float* W_ih = (const float*)d_in[1];
    const float* W_hh = (const float*)d_in[2];
    const float* b_ih = (const float*)d_in[3];
    const float* b_hh = (const float*)d_in[4];
    const float* W_fc = (const float*)d_in[5];
    const float* b_fc = (const float*)d_in[6];

    char* ws = (char*)d_ws;
    _Float16* Whh_p = (_Float16*)(ws + OFF_WHH);
    _Float16* Wih_p = (_Float16*)(ws + OFF_WIH);
    _Float16* Wfc_p = (_Float16*)(ws + OFF_WFC);
    _Float16* hA    = (_Float16*)(ws + OFF_HA);
    _Float16* hB    = (_Float16*)(ws + OFF_HB);
    float*    bias  = (float*)(ws + OFF_BIAS);
    unsigned* ctr   = (unsigned*)(ws + OFF_CTR);
    float*    xp    = (float*)(ws + OFF_XP);

    const size_t xp_bytes = (size_t)SEQ * HID * BATCH * sizeof(float);
    const bool use_xp = ws_size >= (size_t)OFF_XP + xp_bytes;

    // Stage 0: pack weights into WMMA-fragment order (f16) + init state.
    pack_b_frags<<<(HID * HID + 255) / 256, 256, 0, stream>>>(W_hh, Whh_p, HID, HID);
    pack_b_frags<<<(HID * IN_D + 255) / 256, 256, 0, stream>>>(W_ih, Wih_p, HID, IN_D);
    pack_b_frags<<<(OUT_D * HID + 255) / 256, 256, 0, stream>>>(W_fc, Wfc_p, OUT_D, HID);
    init_state<<<(BATCH * HID / 2 + 255) / 256, 256, 0, stream>>>(
        (unsigned*)hA, bias, b_ih, b_hh, ctr);

    if (use_xp) {
        // Stage 1a: fully parallel input projection (keeps the serial loop minimal).
        const int tiles = (SEQ * BATCH / 16) * (HID / 16);
        xproj_gemm<<<tiles / WAVES_PER_WG, TPB, 0, stream>>>(x, Wih_p, bias, xp);
        rnn_persistent<true><<<NUM_WG, TPB, 0, stream>>>(
            x, xp, bias, Whh_p, Wih_p, Wfc_p, b_fc, hA, hB, (float*)d_out, ctr);
    } else {
        rnn_persistent<false><<<NUM_WG, TPB, 0, stream>>>(
            x, xp, bias, Whh_p, Wih_p, Wfc_p, b_fc, hA, hB, (float*)d_out, ctr);
    }
}
